// Transducer_13065290514800
// MI455X (gfx1250) — compile-verified, matrix operand order checked by reading, not run on hardware
//
#include <hip/hip_runtime.h>
#include <hip/hip_bf16.h>

typedef _Float16 h8   __attribute__((ext_vector_type(8)));
typedef _Float16 v16h __attribute__((ext_vector_type(16)));
typedef float    v8f  __attribute__((ext_vector_type(8)));

#define Bn 8
#define Tn 256
#define Un 64
#define U1n 65
#define Vn 512
#define NEG_INF (-1e30f)

// ---------------------------------------------------------------------------
// Kernel 1: per-row max + exp(x - max) stored as f16 (one wave32 per row).
// Rows 0 .. B*T-1 are enc rows, the rest are dec rows.
// ---------------------------------------------------------------------------
__global__ void k_exp_rows(const float* __restrict__ enc,
                           const float* __restrict__ dec,
                           _Float16* __restrict__ Ee,
                           _Float16* __restrict__ De,
                           float* __restrict__ m_enc,
                           float* __restrict__ m_dec) {
  const int lane = threadIdx.x & 31;
  const int wave = threadIdx.x >> 5;
  const int row  = blockIdx.x * (blockDim.x >> 5) + wave;
  const int NE = Bn * Tn;
  const int ND = Bn * U1n;
  if (row >= NE + ND) return;

  const float* src;
  _Float16*    dst;
  float*       mrow;
  if (row < NE) {
    src = enc + (size_t)row * Vn;  dst = Ee + (size_t)row * Vn;  mrow = m_enc + row;
  } else {
    const int r = row - NE;
    src = dec + (size_t)r * Vn;    dst = De + (size_t)r * Vn;    mrow = m_dec + r;
  }

  float v[16];
  float m = -INFINITY;
#pragma unroll
  for (int i = 0; i < 16; ++i) { v[i] = src[lane + i * 32]; m = fmaxf(m, v[i]); }
#pragma unroll
  for (int off = 16; off >= 1; off >>= 1) m = fmaxf(m, __shfl_xor(m, off, 32));
#pragma unroll
  for (int i = 0; i < 16; ++i) dst[lane + i * 32] = (_Float16)__expf(v[i] - m);
  if (lane == 0) *mrow = m;
}

// ---------------------------------------------------------------------------
// Kernel 2: per-wave 16x16 (t,u) tile of S = E_exp @ D_exp^T via
// v_wmma_f32_16x16x32_f16; epilogue computes blank_lp / emit_lp.
//
// A fragment (16-bit A 16x32 layout): lane<16 holds row M=lane with
// K = {kk..kk+7, kk+16..kk+23}; lane>=16 holds row M=lane-16 with
// K = {kk+8..kk+15, kk+24..kk+31}.  B column n = D_exp row (u0+n), so B
// fragments load the same per-lane K pattern from D_exp rows.
// ---------------------------------------------------------------------------
__global__ void k_joint_wmma(const _Float16* __restrict__ Ee,
                             const _Float16* __restrict__ De,
                             const float* __restrict__ m_enc,
                             const float* __restrict__ m_dec,
                             const float* __restrict__ enc,
                             const float* __restrict__ dec,
                             const int* __restrict__ targets,
                             float* __restrict__ blank_lp,
                             float* __restrict__ emit_lp) {
  const int lane = threadIdx.x & 31;
  const int wid  = (blockIdx.x * blockDim.x + threadIdx.x) >> 5;
  const int tu = wid % 5;             // u tile (0..4), last tile masked
  const int tt = (wid / 5) % 16;      // t tile
  const int b  = wid / (5 * 16);
  if (b >= Bn) return;                // uniform per wave

  const int t0 = tt * 16, u0 = tu * 16;
  const int rm = lane & 15;
  const int hi = lane >> 4;

  const int rowA = b * Tn + t0 + rm;
  int uB = u0 + rm; if (uB > Un) uB = Un;     // clamp tail-tile loads
  const int rowB = b * U1n + uB;

  const _Float16* pa = Ee + (size_t)rowA * Vn;
  const _Float16* pb = De + (size_t)rowB * Vn;
  const int koff = hi ? 8 : 0;

  v8f c = {};
#pragma unroll 4
  for (int kk = 0; kk < Vn; kk += 32) {
    v16h a, bb;
    *(h8*)&a        = *(const h8*)(pa + kk + koff);
    *((h8*)&a + 1)  = *(const h8*)(pa + kk + 16 + koff);
    *(h8*)&bb       = *(const h8*)(pb + kk + koff);
    *((h8*)&bb + 1) = *(const h8*)(pb + kk + 16 + koff);
    c = __builtin_amdgcn_wmma_f32_16x16x32_f16(false, a, false, bb,
                                               (short)0, c, false, false);
  }

  // Epilogue: C/D layout -> VGPR r: M = r + 8*hi (t-offset), N = lane%16 (u).
  const int u = u0 + rm;
  if (u >= U1n) return;
  const float md   = m_dec[b * U1n + u];
  const float dec0 = dec[((size_t)(b * U1n + u)) * Vn + 0];
  int tgt = 0; float dtg = 0.f;
  if (u < Un) {
    tgt = targets[b * Un + u];
    if (tgt < 0) tgt = 0; if (tgt >= Vn) tgt = Vn - 1;
    dtg = dec[((size_t)(b * U1n + u)) * Vn + tgt];
  }
#pragma unroll
  for (int r = 0; r < 8; ++r) {
    const int t = t0 + r + hi * 8;
    const float s   = c[r];
    const float lse = __logf(s) + m_enc[b * Tn + t] + md;
    const float e0  = enc[((size_t)(b * Tn + t)) * Vn + 0];
    blank_lp[((size_t)(b * Tn + t)) * U1n + u] = e0 + dec0 - lse;
    if (u < Un) {
      const float et = enc[((size_t)(b * Tn + t)) * Vn + tgt];
      emit_lp[((size_t)(b * Tn + t)) * Un + u] = et + dtg - lse;
    }
  }
}

// ---------------------------------------------------------------------------
// Kernel 3: alpha forward recursion via anti-diagonal wavefront.
// Single block: 8 batches x 68 lanes (65 active). Double-buffered LDS rows.
// alpha[t][u] = logaddexp(alpha[t-1][u] + blank[t-1][u],
//                         alpha[t][u-1] + emit[t][u-1]),  alpha[0][0] = 0.
// ---------------------------------------------------------------------------
__global__ void k_alpha(const float* __restrict__ blank_lp,
                        const float* __restrict__ emit_lp,
                        const int* __restrict__ il,
                        const int* __restrict__ tl,
                        float* __restrict__ out) {
  __shared__ float A0[Bn][68];
  __shared__ float A1[Bn][68];
  __shared__ float nllb[Bn];
  const int tid = threadIdx.x;
  const int b = tid / 68;
  const int u = tid % 68;
  const bool act = (b < Bn) && (u < U1n);
  int ilb = 0, tlb = 0;
  if (act) { ilb = il[b]; tlb = tl[b]; }

  for (int d = 0; d < Tn + U1n - 1; ++d) {
    float*       W = (d & 1) ? &A1[0][0] : &A0[0][0];
    const float* R = (d & 1) ? &A0[0][0] : &A1[0][0];
    if (act) {
      const int t = d - u;
      if (t >= 0 && t < Tn) {
        float top = NEG_INF, left = NEG_INF;
        if (t > 0) top  = R[b * 68 + u]     + blank_lp[((size_t)(b * Tn + t - 1)) * U1n + u];
        if (u > 0) left = R[b * 68 + u - 1] + emit_lp [((size_t)(b * Tn + t)) * Un + (u - 1)];
        float val;
        if (d == 0) {
          val = 0.f;
        } else {
          const float mx = fmaxf(top, left);
          val = mx + __logf(__expf(top - mx) + __expf(left - mx));
        }
        W[b * 68 + u] = val;
        if (t == ilb - 1 && u == tlb)
          nllb[b] = -(val + blank_lp[((size_t)(b * Tn + t)) * U1n + u]);
      }
    }
    __syncthreads();
  }

  if (tid == 0) {
    float s = 0.f;
    for (int i = 0; i < Bn; ++i) s += nllb[i];
    out[0] = s / (float)Bn;
  }
}

// ---------------------------------------------------------------------------
extern "C" void kernel_launch(void* const* d_in, const int* in_sizes, int n_in,
                              void* d_out, int out_size, void* d_ws, size_t ws_size,
                              hipStream_t stream) {
  const float* enc     = (const float*)d_in[0];
  const float* dec     = (const float*)d_in[1];
  const int*   targets = (const int*)d_in[2];
  const int*   il      = (const int*)d_in[3];
  const int*   tl      = (const int*)d_in[4];
  float* out = (float*)d_out;

  char* ws = (char*)d_ws;
  size_t off = 0;
  _Float16* Ee = (_Float16*)(ws + off); off += (size_t)Bn * Tn  * Vn * sizeof(_Float16);
  _Float16* De = (_Float16*)(ws + off); off += (size_t)Bn * U1n * Vn * sizeof(_Float16);
  float* m_enc = (float*)(ws + off);    off += (size_t)Bn * Tn  * sizeof(float);
  float* m_dec = (float*)(ws + off);    off += (size_t)Bn * U1n * sizeof(float);
  float* blank = (float*)(ws + off);    off += (size_t)Bn * Tn * U1n * sizeof(float);
  float* emit  = (float*)(ws + off);    off += (size_t)Bn * Tn * Un  * sizeof(float);

  // Kernel 1: 2568 rows, 8 waves (rows) per 256-thread block.
  const int rows = Bn * Tn + Bn * U1n;
  hipLaunchKernelGGL(k_exp_rows, dim3((rows + 7) / 8), dim3(256), 0, stream,
                     enc, dec, Ee, De, m_enc, m_dec);

  // Kernel 2: 8 batches * 16 t-tiles * 5 u-tiles = 640 waves -> 80 blocks of 8 waves.
  hipLaunchKernelGGL(k_joint_wmma, dim3(80), dim3(256), 0, stream,
                     Ee, De, m_enc, m_dec, enc, dec, targets, blank, emit);

  // Kernel 3: single block wavefront (8 batches x 68 lanes).
  hipLaunchKernelGGL(k_alpha, dim3(1), dim3(544), 0, stream,
                     blank, emit, il, tl, out);
}
